// VectorQuantize_44719199486120
// MI455X (gfx1250) — matrix operand order repaired and checked by reading, not compiled, not run on hardware
//
#include <hip/hip_runtime.h>
#include <hip/hip_bf16.h>
#include <stdint.h>

typedef __attribute__((ext_vector_type(16))) __bf16 v16bf;
typedef __attribute__((ext_vector_type(8)))  __bf16 v8bf;
typedef __attribute__((ext_vector_type(8)))  float  v8f;

#define DIM      256
#define KCH      8          // DIM / 32 K-chunks per WMMA pass
#define NROWS_X  32768      // 8 * 4096
#define NROWS_E  8192
#define NTILES   (NROWS_E / 16)

// ---------------------------------------------------------------------------
// Zero the single loss accumulator (harness poisons d_out with 0xAA).
// ---------------------------------------------------------------------------
__global__ void vq_zero_loss(float* loss) { *loss = 0.0f; }

// ---------------------------------------------------------------------------
// Split f32 rows into bf16 hi/lo pairs; optionally emit per-row ||e||^2.
// One wave per row; 8 coalesced f32 loads per lane.
// ---------------------------------------------------------------------------
__global__ __launch_bounds__(256) void vq_split_rows(
    const float* __restrict__ src, __bf16* __restrict__ hi,
    __bf16* __restrict__ lo, float* __restrict__ norm, int nrows)
{
    const int wave = (int)((blockIdx.x * blockDim.x + threadIdx.x) >> 5);
    const int lane = (int)(threadIdx.x & 31);
    if (wave >= nrows) return;
    const float* r = src + (size_t)wave * DIM;
    float s = 0.0f;
#pragma unroll
    for (int j = 0; j < 8; ++j) {
        const int k = lane + j * 32;
        const float v = r[k];
        const __bf16 h = (__bf16)v;
        const float  res = v - (float)h;
        hi[(size_t)wave * DIM + k] = h;
        lo[(size_t)wave * DIM + k] = (__bf16)res;
        s += v * v;
    }
    if (norm) {
#pragma unroll
        for (int m = 16; m >= 1; m >>= 1) s += __shfl_xor(s, m, 32);
        if (lane == 0) norm[wave] = s;
    }
}

// ---------------------------------------------------------------------------
// CDNA5 async memory->LDS staging (no VGPR data path, tracked by ASYNCcnt).
// LDS address = low 32 bits of flat shared pointer (aperture truncation rule).
// ---------------------------------------------------------------------------
__device__ __forceinline__ void async_b128_to_lds(const void* gaddr, const void* lds)
{
    const uint32_t la = (uint32_t)(uintptr_t)lds;
    asm volatile("global_load_async_to_lds_b128 %0, %1, off"
                 :: "v"(la), "v"(gaddr) : "memory");
}

__device__ __forceinline__ void wait_asynccnt0()
{
    asm volatile("s_wait_asynccnt 0" ::: "memory");
}

// Stage one codebook tile (16 rows x 256 bf16, hi+lo = 16KB) into LDS:
// 1024 16-byte chunks across 256 threads, 4 async b128 issues per thread.
__device__ __forceinline__ void stage_async(const __bf16* __restrict__ eh,
                                            const __bf16* __restrict__ el,
                                            int n0, int t,
                                            const __bf16* dh, const __bf16* dl)
{
#pragma unroll
    for (int j = 0; j < 4; ++j) {
        const int c = t * 4 + j;
        const __bf16* src = (c < 512) ? eh : el;
        const __bf16* dst = (c < 512) ? dh : dl;
        const int cc = c & 511;
        async_b128_to_lds(src + (size_t)n0 * DIM + cc * 8, dst + cc * 8);
    }
}

// ---------------------------------------------------------------------------
// Fused distance GEMM + running argmax.
// score(row,n) = 2 * dot(x_row, e_n) - ||e_n||^2  (||x||^2 dropped: constant)
// dot computed as bf16x3: hi*hi + hi*lo + lo*hi with f32 WMMA accumulate.
// Block = 256 threads = 8 waves; wave = one 16-row A tile; 512 N tiles.
// ---------------------------------------------------------------------------
__global__ __launch_bounds__(256) void vq_argmax(
    const __bf16* __restrict__ xh, const __bf16* __restrict__ xl,
    const __bf16* __restrict__ eh, const __bf16* __restrict__ el,
    const float* __restrict__ enorm, int* __restrict__ bestidx)
{
    __shared__ __align__(32) __bf16 Bh[2][16 * DIM];
    __shared__ __align__(32) __bf16 Bl[2][16 * DIM];

    const int t  = (int)threadIdx.x;
    const int wv = t >> 5;
    const int ln = t & 31;
    const int m  = ln & 15;      // A: row-in-tile / B,C: column N-in-tile
    const int h  = ln >> 4;      // lane half selector
    const int row0 = ((int)blockIdx.x * 8 + wv) * 16;

    // --- A operands resident in VGPRs, contiguous K order (not the ISA table order):
    // lane 16h+m holds row M=m with K = 16h..16h+15 (matches B order)
    v16bf ah[KCH], al[KCH];
    {
        const __bf16* ph = xh + (size_t)(row0 + m) * DIM;
        const __bf16* pl = xl + (size_t)(row0 + m) * DIM;
#pragma unroll
        for (int kc = 0; kc < KCH; ++kc) {
            const int o1 = kc * 32 + h * 16;
            const int o2 = kc * 32 + h * 16 + 8;
            const v8bf a0 = *(const v8bf*)(ph + o1);
            const v8bf a1 = *(const v8bf*)(ph + o2);
            const v8bf b0 = *(const v8bf*)(pl + o1);
            const v8bf b1 = *(const v8bf*)(pl + o2);
#pragma unroll
            for (int i = 0; i < 8; ++i) {
                ah[kc][i] = a0[i]; ah[kc][8 + i] = a1[i];
                al[kc][i] = b0[i]; al[kc][8 + i] = b1[i];
            }
        }
    }

    float best[8];
    int   bidx[8];
#pragma unroll
    for (int r = 0; r < 8; ++r) { best[r] = -3.402823466e38f; bidx[r] = 0; }

    // Prime buffer 0 via async DMA, then enter the software pipeline.
    stage_async(eh, el, 0, t, Bh[0], Bl[0]);
    wait_asynccnt0();
    __syncthreads();

    for (int nt = 0; nt < NTILES; ++nt) {
        const int buf = nt & 1;
        // Kick the DMA for the next tile into the other buffer (the barrier at
        // the end of the previous iteration guarantees nobody still reads it).
        if (nt + 1 < NTILES)
            stage_async(eh, el, (nt + 1) * 16, t, Bh[buf ^ 1], Bl[buf ^ 1]);

        // B operand, contiguous K order (same as A): lane = 16h + n holds column n,
        // K = kc*32 + 16h .. +15 (32 contiguous bytes per lane).
        const __bf16* pbh = &Bh[buf][m * DIM + h * 16];
        const __bf16* pbl = &Bl[buf][m * DIM + h * 16];

        v8f acc = {};
#pragma unroll
        for (int kc = 0; kc < KCH; ++kc) {
            const v16bf bh = *(const v16bf*)(pbh + kc * 32);
            const v16bf bl = *(const v16bf*)(pbl + kc * 32);
            acc = __builtin_amdgcn_wmma_f32_16x16x32_bf16(false, ah[kc], false, bh,
                                                          (short)0, acc, false, false);
            acc = __builtin_amdgcn_wmma_f32_16x16x32_bf16(false, ah[kc], false, bl,
                                                          (short)0, acc, false, false);
            acc = __builtin_amdgcn_wmma_f32_16x16x32_bf16(false, al[kc], false, bh,
                                                          (short)0, acc, false, false);
        }

        const int   n  = nt * 16 + m;
        const float en = enorm[n];
#pragma unroll
        for (int r = 0; r < 8; ++r) {          // C layout: M = r + 8h, N = m
            const float s = 2.0f * acc[r] - en;
            if (s > best[r]) { best[r] = s; bidx[r] = n; }
        }

        // Our async transfers must land before anyone reads the other buffer.
        if (nt + 1 < NTILES) wait_asynccnt0();
        __syncthreads();
    }

    // Cross-lane argmax over the 16 N-candidates per row (halves independent).
#pragma unroll
    for (int r = 0; r < 8; ++r) {
        float s = best[r];
        int   i = bidx[r];
#pragma unroll
        for (int msk = 8; msk >= 1; msk >>= 1) {
            const float os = __shfl_xor(s, msk, 32);
            const int   oi = __shfl_xor(i, msk, 32);
            if (os > s || (os == s && oi < i)) { s = os; i = oi; }
        }
        if (m == 0) bestidx[row0 + r + 8 * h] = i;
    }
}

// ---------------------------------------------------------------------------
// Gather exact f32 codebook rows, write indices (as float), accumulate loss.
// One wave per row; 2x float4 per lane.
// ---------------------------------------------------------------------------
__global__ __launch_bounds__(256) void vq_gather_loss(
    const float* __restrict__ x, const float* __restrict__ embed,
    const int* __restrict__ bestidx, float* __restrict__ out_q,
    float* __restrict__ out_ind, float* __restrict__ out_loss)
{
    const int row  = (int)((blockIdx.x * blockDim.x + threadIdx.x) >> 5);
    const int lane = (int)(threadIdx.x & 31);
    if (row >= NROWS_X) return;

    const int k = bestidx[row];
    const float* e  = embed + (size_t)k * DIM;
    const float* xr = x + (size_t)row * DIM;
    float* q = out_q + (size_t)row * DIM;

    float s = 0.0f;
#pragma unroll
    for (int j = 0; j < 2; ++j) {
        const int o = lane * 4 + j * 128;
        const float4 ev = *(const float4*)(e + o);
        const float4 xv = *(const float4*)(xr + o);
        *(float4*)(q + o) = ev;
        const float dx = ev.x - xv.x, dy = ev.y - xv.y;
        const float dz = ev.z - xv.z, dw = ev.w - xv.w;
        s += dx * dx + dy * dy + dz * dz + dw * dw;
    }
#pragma unroll
    for (int m = 16; m >= 1; m >>= 1) s += __shfl_xor(s, m, 32);
    if (lane == 0) {
        out_ind[row] = (float)k;
        atomicAdd(out_loss, s * (1.0f / (float)((size_t)NROWS_X * DIM)));
    }
}

// ---------------------------------------------------------------------------
extern "C" void kernel_launch(void* const* d_in, const int* in_sizes, int n_in,
                              void* d_out, int out_size, void* d_ws, size_t ws_size,
                              hipStream_t stream)
{
    const float* x     = (const float*)d_in[0];   // [8,4096,256]
    const float* embed = (const float*)d_in[1];   // [8192,256]

    float* out      = (float*)d_out;
    float* out_q    = out;                               // 8,388,608 floats
    float* out_ind  = out + (size_t)NROWS_X * DIM;       // 32,768 floats
    float* out_loss = out_ind + NROWS_X;                 // 1 float

    char*  ws  = (char*)d_ws;
    size_t off = 0;
    auto carve = [&](size_t bytes) -> char* {
        char* p = ws + off;
        off += (bytes + 255) & ~(size_t)255;
        return p;
    };
    __bf16* xh = (__bf16*)carve((size_t)NROWS_X * DIM * 2);
    __bf16* xl = (__bf16*)carve((size_t)NROWS_X * DIM * 2);
    __bf16* eh = (__bf16*)carve((size_t)NROWS_E * DIM * 2);
    __bf16* el = (__bf16*)carve((size_t)NROWS_E * DIM * 2);
    float*  en = (float*)carve((size_t)NROWS_E * 4);
    int*    bi = (int*)carve((size_t)NROWS_X * 4);

    vq_zero_loss<<<1, 1, 0, stream>>>(out_loss);
    vq_split_rows<<<NROWS_E / 8, 256, 0, stream>>>(embed, eh, el, en, NROWS_E);
    vq_split_rows<<<NROWS_X / 8, 256, 0, stream>>>(x, xh, xl, nullptr, NROWS_X);
    vq_argmax<<<NROWS_X / 128, 256, 0, stream>>>(xh, xl, eh, el, en, bi);
    vq_gather_loss<<<NROWS_X / 8, 256, 0, stream>>>(x, embed, bi, out_q, out_ind, out_loss);
}
